// STVN_65910568124560
// MI455X (gfx1250) — compile-verified, matrix operand order checked by reading, not run on hardware
//
#include <hip/hip_runtime.h>

#define NTOT 23552
#define NPG  23
#define NG   1024
#define NB   64
#define NT_  16
#define HID  48
#define NHEAD 4
#define NGRU 64

typedef __attribute__((ext_vector_type(16))) __bf16 v16bf;
typedef __attribute__((ext_vector_type(8)))  float  v8f;

__device__ __forceinline__ unsigned short f2bf(float f) {
  union { float f; unsigned u; } c; c.f = f;
  return (unsigned short)((c.u + 0x7FFFu + ((c.u >> 16) & 1u)) >> 16);
}
__device__ __forceinline__ float bf2f(unsigned short s) {
  union { unsigned u; float f; } c; c.u = ((unsigned)s) << 16;
  return c.f;
}

// Bijective swizzle (row-in-tile, k-in-32) -> linear index (lane*16 + elem) matching
// the ISA 16-bit A/B fragment layout: K bit4 -> VGPR bit2; K bit3 -> lane group;
// K bits2:1 -> VGPR bits1:0; K bit0 -> 16-bit half. Same formula for A (r=M) and B (r=N).
__device__ __forceinline__ int swz(int r, int kk) {
  int lane = r + (((kk >> 3) & 1) << 4);
  int v = (((kk >> 4) & 1) << 2) | ((kk >> 1) & 3);
  return lane * 16 + v * 2 + (kk & 1);
}

// Fragment load from a pre-swizzled 512-element bf16 tile: 2 x b128 per lane.
__device__ __forceinline__ v16bf load_frag(const unsigned short* base, int lane) {
  union { v16bf v; uint4 q[2]; } f;
  const uint4* p = (const uint4*)(base + (lane << 4));
  f.q[0] = p[0];
  f.q[1] = p[1];
  return f.v;
}

// ---------------- Tensor Data Mover: stage a 2D bf16 tile into LDS ----------------
#if defined(__gfx1250__) && __has_builtin(__builtin_amdgcn_tensor_load_to_lds) && __has_builtin(__builtin_amdgcn_s_wait_tensorcnt)
#define HAVE_TDM 1
typedef __attribute__((ext_vector_type(4))) unsigned int tdmu4;
typedef __attribute__((ext_vector_type(8))) int tdmi8;
typedef __attribute__((ext_vector_type(4))) int tdmi4;
__device__ __forceinline__ void tdm_load_2d(unsigned lds_off, const void* gptr,
                                            unsigned rowElems, unsigned rows,
                                            unsigned strideElems) {
  unsigned long long ga = (unsigned long long)(uintptr_t)gptr;
  tdmu4 g0;
  g0[0] = 1u;                                               // count=1, user desc
  g0[1] = lds_off;                                          // lds_addr (bytes)
  g0[2] = (unsigned)ga;                                     // global_addr[31:0]
  g0[3] = ((unsigned)(ga >> 32) & 0x01FFFFFFu) | 0x80000000u; // addr[56:32] | type=2
  tdmi8 g1;
  g1[0] = 0x00010000;                 // workgroup_mask=0, data_size=1 (2 bytes)
  g1[1] = (int)(rowElems << 16);      // tensor_dim0[15:0] in [63:48]
  g1[2] = (int)(rows << 16);          // tensor_dim0 hi=0 | tensor_dim1[15:0]
  g1[3] = (int)(rowElems << 16);      // tensor_dim1 hi=0 | tile_dim0
  g1[4] = (int)rows;                  // tile_dim1 | tile_dim2=0
  g1[5] = (int)strideElems;           // tensor_dim0_stride[31:0]
  g1[6] = 0;
  g1[7] = 0;
  tdmi4 gz = {0, 0, 0, 0};
#if __clang_major__ >= 23
  tdmi8 gz8 = {0, 0, 0, 0, 0, 0, 0, 0};
  __builtin_amdgcn_tensor_load_to_lds(g0, g1, gz, gz, gz8, 0);
#else
  __builtin_amdgcn_tensor_load_to_lds(g0, g1, gz, gz, 0);
#endif
  __builtin_amdgcn_s_wait_tensorcnt(0);
}
#endif

// ---------------- prep kernels: bf16 convert + fragment swizzle ----------------
__global__ void prep_x(const float* __restrict__ x, unsigned short* __restrict__ xb) {
  int i = blockIdx.x * 256 + threadIdx.x;
  if (i >= NTOT * 32) return;
  int n = i >> 5, kk = i & 31;
  unsigned short v = (kk < 11) ? f2bf(x[n * 11 + kk]) : (unsigned short)0;
  xb[(size_t)(n >> 4) * 512 + swz(n & 15, kk)] = v;        // [mt][512]
}
__global__ void prep_w1(const float* __restrict__ W1, unsigned short* __restrict__ w1b) {
  int i = blockIdx.x * 256 + threadIdx.x;
  if (i >= 32 * 192) return;
  int k = i / 192, n = i % 192;
  unsigned short v = (k < 11) ? f2bf(W1[k * 192 + n]) : (unsigned short)0;
  w1b[(size_t)(n >> 4) * 512 + swz(n & 15, k)] = v;        // [nt][512], kt==0
}
__global__ void prep_w2(const float* __restrict__ W2, unsigned short* __restrict__ w2b) {
  int i = blockIdx.x * 256 + threadIdx.x;
  if (i >= 192 * 48) return;
  int k = i / 48, n = i % 48;
  w2b[((size_t)(k >> 5) * 3 + (n >> 4)) * 512 + swz(n & 15, k & 31)] = f2bf(W2[i]);
}
// W is (192,K) row-major; emit swizzled W^T tiles: [kt][nt(12)][512]
__global__ void prep_wT(const float* __restrict__ W, unsigned short* __restrict__ out, int K) {
  int i = blockIdx.x * 256 + threadIdx.x;
  if (i >= K * 192) return;
  int k = i / 192, n = i % 192;
  out[((size_t)(k >> 5) * 12 + (n >> 4)) * 512 + swz(n & 15, k & 31)] = f2bf(W[n * K + k]);
}

// ---------------- GEMM1: h1 = x @ W1  (23552x32pad @ 32x192) ----------------
__global__ void gemm1(const unsigned short* __restrict__ xb,
                      const unsigned short* __restrict__ w1b,
                      unsigned short* __restrict__ h1b) {
  int lane = threadIdx.x & 31, wave = threadIdx.x >> 5;
  int tile = blockIdx.x * 8 + wave;            // 1472*12 tiles total
  int mt = tile / 12, nt = tile % 12;
  v16bf a = load_frag(xb + (size_t)mt * 512, lane);
  v16bf b = load_frag(w1b + (size_t)nt * 512, lane);
  v8f c = {};
  c = __builtin_amdgcn_wmma_f32_16x16x32_bf16(false, a, false, b, (short)0, c, false, false);
  int n = nt * 16 + (lane & 15), mb = (lane >> 4) << 3;
#pragma unroll
  for (int v = 0; v < 8; ++v)
    h1b[(size_t)(mt * 16 + mb + v) * 192 + n] = f2bf(c[v]);
}

// ---------------- attention logits, layer 1 ----------------
__global__ void al1_kernel(const unsigned short* __restrict__ h1b,
                           const float* __restrict__ as1, const float* __restrict__ ad1,
                           float* __restrict__ als, float* __restrict__ ald) {
  int i = blockIdx.x * 256 + threadIdx.x;
  if (i >= NTOT * NHEAD) return;
  int n = i >> 2, h = i & 3;
  const unsigned short* hp = h1b + (size_t)n * 192 + h * HID;
  float s = 0.f, d = 0.f;
#pragma unroll
  for (int k = 0; k < HID; ++k) {
    float v = bf2f(hp[k]);
    s += v * as1[h * HID + k];
    d += v * ad1[h * HID + k];
  }
  als[i] = s; ald[i] = d;
}

// ---------------- per-graph softmax + aggregate + bias + ELU (layer 1) ----------------
__global__ void agg1(const unsigned short* __restrict__ h1b,
                     const float* __restrict__ alsG, const float* __restrict__ aldG,
                     const float* __restrict__ b1,
                     unsigned short* __restrict__ act1b) {
  __shared__ float als[NPG * NHEAD], ald[NPG * NHEAD], alpha[NPG * NPG * NHEAD];
  __shared__ __align__(16) unsigned short tileh[NPG * 192];   // 8832 B, TDM-staged
  int g = blockIdx.x, tid = threadIdx.x, base = g * NPG;
#ifdef HAVE_TDM
  if (tid < 32)
    tdm_load_2d((unsigned)(uintptr_t)tileh, h1b + (size_t)base * 192, 192, NPG, 192);
#else
  for (int i = tid; i < NPG * 192; i += 256) tileh[i] = h1b[(size_t)base * 192 + i];
#endif
  for (int i = tid; i < NPG * NHEAD; i += 256) { als[i] = alsG[base * NHEAD + i]; ald[i] = aldG[base * NHEAD + i]; }
  __syncthreads();
  for (int t = tid; t < NPG * NHEAD; t += 256) {   // one (dst,head) pair per thread
    int d = t >> 2, h = t & 3;
    float ev[NPG], mx = -3.0e38f;
#pragma unroll
    for (int s = 0; s < NPG; ++s) {
      float e = als[s * NHEAD + h] + ald[d * NHEAD + h];
      e = e > 0.f ? e : 0.2f * e;                  // leaky_relu
      ev[s] = e; mx = fmaxf(mx, e);
    }
    float sum = 0.f;
#pragma unroll
    for (int s = 0; s < NPG; ++s) { float w = __expf(ev[s] - mx); alpha[(s * NPG + d) * NHEAD + h] = w; sum += w; }
    float inv = 1.f / (sum + 1e-16f);
#pragma unroll
    for (int s = 0; s < NPG; ++s) alpha[(s * NPG + d) * NHEAD + h] *= inv;
  }
  __syncthreads();
  for (int o = tid; o < NPG * 192; o += 256) {
    int d = o / 192, f = o % 192, h = f / HID;
    float acc = 0.f;
#pragma unroll
    for (int s = 0; s < NPG; ++s)
      acc += alpha[(s * NPG + d) * NHEAD + h] * bf2f(tileh[s * 192 + f]);
    acc += b1[f];
    acc = acc > 0.f ? acc : (__expf(acc) - 1.f);   // ELU
    int m = base + d;                               // swizzled store for GEMM2
    act1b[((size_t)(m >> 4) * 6 + (f >> 5)) * 512 + swz(m & 15, f & 31)] = f2bf(acc);
  }
}

// ---------------- GEMM2: h2 = act1 @ W2  (23552x192 @ 192x48) ----------------
__global__ void gemm2(const unsigned short* __restrict__ act1b,
                      const unsigned short* __restrict__ w2b,
                      unsigned short* __restrict__ h2b) {
  int lane = threadIdx.x & 31, wave = threadIdx.x >> 5;
  int tile = blockIdx.x * 8 + wave;            // 1472*3 tiles total
  int mt = tile / 3, nt = tile % 3;
  v8f c = {};
#pragma unroll
  for (int kt = 0; kt < 6; ++kt) {
    if (kt < 5)
      __builtin_prefetch(act1b + ((size_t)mt * 6 + kt + 1) * 512 + (lane << 4), 0, 0);
    v16bf a = load_frag(act1b + ((size_t)mt * 6 + kt) * 512, lane);
    v16bf b = load_frag(w2b + ((size_t)kt * 3 + nt) * 512, lane);
    c = __builtin_amdgcn_wmma_f32_16x16x32_bf16(false, a, false, b, (short)0, c, false, false);
  }
  int n = nt * 16 + (lane & 15), mb = (lane >> 4) << 3;
#pragma unroll
  for (int v = 0; v < 8; ++v)
    h2b[(size_t)(mt * 16 + mb + v) * 48 + n] = f2bf(c[v]);
}

// ---------------- attention logits, layer 2 (1 head) ----------------
__global__ void al2_kernel(const unsigned short* __restrict__ h2b,
                           const float* __restrict__ as2, const float* __restrict__ ad2,
                           float* __restrict__ als, float* __restrict__ ald) {
  int n = blockIdx.x * 256 + threadIdx.x;
  if (n >= NTOT) return;
  const unsigned short* hp = h2b + (size_t)n * 48;
  float s = 0.f, d = 0.f;
#pragma unroll
  for (int k = 0; k < HID; ++k) { float v = bf2f(hp[k]); s += v * as2[k]; d += v * ad2[k]; }
  als[n] = s; ald[n] = d;
}

// ---------------- per-graph softmax + aggregate + ELU + mean/max pool (layer 2) ----------------
__global__ void agg2_pool(const unsigned short* __restrict__ h2b,
                          const float* __restrict__ alsG, const float* __restrict__ aldG,
                          const float* __restrict__ b2,
                          unsigned short* __restrict__ framesb) {
  __shared__ float als[NPG], ald[NPG], alpha[NPG * NPG], hh[NPG * HID];
  __shared__ __align__(16) unsigned short tileh[NPG * 48];    // 2208 B, TDM-staged
  int g = blockIdx.x, tid = threadIdx.x, base = g * NPG;
#ifdef HAVE_TDM
  if (tid < 32)
    tdm_load_2d((unsigned)(uintptr_t)tileh, h2b + (size_t)base * 48, 48, NPG, 48);
#else
  for (int i = tid; i < NPG * 48; i += 256) tileh[i] = h2b[(size_t)base * 48 + i];
#endif
  if (tid < NPG) { als[tid] = alsG[base + tid]; ald[tid] = aldG[base + tid]; }
  __syncthreads();
  if (tid < NPG) {                                  // thread = dst node
    int d = tid;
    float ev[NPG], mx = -3.0e38f;
#pragma unroll
    for (int s = 0; s < NPG; ++s) {
      float e = als[s] + ald[d];
      e = e > 0.f ? e : 0.2f * e;
      ev[s] = e; mx = fmaxf(mx, e);
    }
    float sum = 0.f;
#pragma unroll
    for (int s = 0; s < NPG; ++s) { float w = __expf(ev[s] - mx); alpha[s * NPG + d] = w; sum += w; }
    float inv = 1.f / (sum + 1e-16f);
#pragma unroll
    for (int s = 0; s < NPG; ++s) alpha[s * NPG + d] *= inv;
  }
  __syncthreads();
  for (int o = tid; o < NPG * HID; o += 256) {
    int d = o / HID, k = o % HID;
    float acc = 0.f;
#pragma unroll
    for (int s = 0; s < NPG; ++s)
      acc += alpha[s * NPG + d] * bf2f(tileh[s * 48 + k]);
    acc += b2[k];
    acc = acc > 0.f ? acc : (__expf(acc) - 1.f);
    hh[d * HID + k] = acc;
  }
  __syncthreads();
  if (tid < HID) {
    int k = tid;
    float mean = 0.f, mx = -3.0e38f;
#pragma unroll
    for (int d = 0; d < NPG; ++d) { float v = hh[d * HID + k]; mean += v; mx = fmaxf(mx, v); }
    mean *= (1.0f / NPG);
    int b = g >> 4, t = g & 15;                     // g = b*T + t
    // swizzled frame store: [t][mt(4)][kt(3)][512]
    framesb[((size_t)(t * 4 + (b >> 4)) * 3 + (k >> 5)) * 512 + swz(b & 15, k & 31)] = f2bf(mean);
    int k2 = 48 + k;
    framesb[((size_t)(t * 4 + (b >> 4)) * 3 + (k2 >> 5)) * 512 + swz(b & 15, k2 & 31)] = f2bf(mx);
  }
}

// ---------------- persistent single-WG WMMA GRU layer ----------------
// xsw: swizzled [t][mt(4)][kt(ktn)][512]; wib: [kt][nt(12)][512]; whb: [kt(2)][nt(12)][512]
__global__ void gru_layer(const unsigned short* __restrict__ xsw, int ktn,
                          const unsigned short* __restrict__ wib,
                          const unsigned short* __restrict__ whb,
                          const float* __restrict__ bih, const float* __restrict__ bhh,
                          unsigned short* __restrict__ outsw,    // [t][mt(4)][kt(2)][512] or null
                          float* __restrict__ lastout) {         // (B,64) f32 or null
  __shared__ __align__(16) unsigned short hb[8 * 512];  // hidden state, swizzled [mt4][kt2][512]
  __shared__ unsigned short gi[NB * 192];               // input gates (+bih)
  __shared__ unsigned short gh[NB * 192];               // hidden gates (+bhh)
  int tid = threadIdx.x, lane = tid & 31, wave = tid >> 5;
  for (int i = tid; i < 8 * 512; i += 256) hb[i] = 0;
  __syncthreads();
  for (int t = 0; t < NT_; ++t) {
    for (int tt = wave * 6; tt < wave * 6 + 6; ++tt) {  // 4 x 12 = 48 tiles, 6 per wave
      int mt = tt / 12, nt = tt % 12;
      int m0 = mt * 16;
      v8f c = {};
      for (int kt = 0; kt < ktn; ++kt) {
        v16bf a = load_frag(xsw + ((size_t)(t * 4 + mt) * ktn + kt) * 512, lane);
        v16bf b = load_frag(wib + ((size_t)kt * 12 + nt) * 512, lane);
        c = __builtin_amdgcn_wmma_f32_16x16x32_bf16(false, a, false, b, (short)0, c, false, false);
      }
      v8f d = {};
#pragma unroll
      for (int kt = 0; kt < 2; ++kt) {
        v16bf a = load_frag(hb + ((size_t)mt * 2 + kt) * 512, lane);   // ds_load_b128 x2
        v16bf b = load_frag(whb + ((size_t)kt * 12 + nt) * 512, lane);
        d = __builtin_amdgcn_wmma_f32_16x16x32_bf16(false, a, false, b, (short)0, d, false, false);
      }
      int n = nt * 16 + (lane & 15), mb = (lane >> 4) << 3;
#pragma unroll
      for (int v = 0; v < 8; ++v) {
        gi[(m0 + mb + v) * 192 + n] = f2bf(c[v] + bih[n]);
        gh[(m0 + mb + v) * 192 + n] = f2bf(d[v] + bhh[n]);
      }
    }
    __syncthreads();
    for (int i = tid; i < NB * NGRU; i += 256) {
      int b = i >> 6, j = i & 63;
      float ir = bf2f(gi[b * 192 + j]),        hr = bf2f(gh[b * 192 + j]);
      float iz = bf2f(gi[b * 192 + 64 + j]),   hz = bf2f(gh[b * 192 + 64 + j]);
      float in_ = bf2f(gi[b * 192 + 128 + j]), hn = bf2f(gh[b * 192 + 128 + j]);
      float r = 1.f / (1.f + __expf(-(ir + hr)));
      float z = 1.f / (1.f + __expf(-(iz + hz)));
      float nn = tanhf(in_ + r * hn);
      int hidx = (((b >> 4) << 1) + (j >> 5)) * 512 + swz(b & 15, j & 31);
      float h2 = (1.f - z) * nn + z * bf2f(hb[hidx]);
      hb[hidx] = f2bf(h2);
      if (outsw)
        outsw[((size_t)(t * 4 + (b >> 4)) * 2 + (j >> 5)) * 512 + swz(b & 15, j & 31)] = f2bf(h2);
      if (lastout && t == NT_ - 1) lastout[i] = h2;
    }
    __syncthreads();
  }
}

// ---------------- MLP head ----------------
__global__ void head_kernel(const float* __restrict__ last,
                            const float* __restrict__ Wh1, const float* __restrict__ bh1,
                            const float* __restrict__ Wh2, const float* __restrict__ bh2,
                            float* __restrict__ out) {
  int b = threadIdx.x;
  if (b >= NB) return;
  float acc = bh2[0];
#pragma unroll 4
  for (int j = 0; j < 32; ++j) {
    float s = bh1[j];
    for (int k = 0; k < NGRU; ++k) s += last[b * NGRU + k] * Wh1[k * 32 + j];
    acc += fmaxf(s, 0.f) * Wh2[j];
  }
  out[b] = acc;
}

extern "C" void kernel_launch(void* const* d_in, const int* in_sizes, int n_in,
                              void* d_out, int out_size, void* d_ws, size_t ws_size,
                              hipStream_t stream) {
  const float* x    = (const float*)d_in[0];
  const float* W1   = (const float*)d_in[3];
  const float* as1  = (const float*)d_in[4];
  const float* ad1  = (const float*)d_in[5];
  const float* b1   = (const float*)d_in[6];
  const float* W2   = (const float*)d_in[7];
  const float* as2  = (const float*)d_in[8];
  const float* ad2  = (const float*)d_in[9];
  const float* b2   = (const float*)d_in[10];
  const float* Wih0 = (const float*)d_in[11];
  const float* Whh0 = (const float*)d_in[12];
  const float* bih0 = (const float*)d_in[13];
  const float* bhh0 = (const float*)d_in[14];
  const float* Wih1 = (const float*)d_in[15];
  const float* Whh1 = (const float*)d_in[16];
  const float* bih1 = (const float*)d_in[17];
  const float* bhh1 = (const float*)d_in[18];
  const float* Wh1  = (const float*)d_in[19];
  const float* bh1  = (const float*)d_in[20];
  const float* Wh2  = (const float*)d_in[21];
  const float* bh2  = (const float*)d_in[22];
  float* out = (float*)d_out;

  char* p = (char*)d_ws;
  auto alloc = [&](size_t bytes) -> void* {
    void* r = (void*)p;
    p += (bytes + 255) & ~(size_t)255;
    return r;
  };
  unsigned short* xb     = (unsigned short*)alloc((size_t)1472 * 512 * 2);
  unsigned short* w1b    = (unsigned short*)alloc((size_t)12 * 512 * 2);
  unsigned short* h1b    = (unsigned short*)alloc((size_t)NTOT * 192 * 2);
  float*          al1s   = (float*)alloc((size_t)NTOT * 4 * 4);
  float*          al1d   = (float*)alloc((size_t)NTOT * 4 * 4);
  unsigned short* act1b  = (unsigned short*)alloc((size_t)1472 * 6 * 512 * 2);
  unsigned short* w2b    = (unsigned short*)alloc((size_t)6 * 3 * 512 * 2);
  unsigned short* h2b    = (unsigned short*)alloc((size_t)NTOT * 48 * 2);
  float*          al2s   = (float*)alloc((size_t)NTOT * 4);
  float*          al2d   = (float*)alloc((size_t)NTOT * 4);
  unsigned short* framesb= (unsigned short*)alloc((size_t)NT_ * 4 * 3 * 512 * 2);
  unsigned short* wih0b  = (unsigned short*)alloc((size_t)3 * 12 * 512 * 2);
  unsigned short* whh0b  = (unsigned short*)alloc((size_t)2 * 12 * 512 * 2);
  unsigned short* wih1b  = (unsigned short*)alloc((size_t)2 * 12 * 512 * 2);
  unsigned short* whh1b  = (unsigned short*)alloc((size_t)2 * 12 * 512 * 2);
  unsigned short* l0seq  = (unsigned short*)alloc((size_t)NT_ * 4 * 2 * 512 * 2);
  float*          last   = (float*)alloc(NB * NGRU * 4);

  // prep: bf16 conversion + fragment swizzle
  prep_x<<<(NTOT * 32 + 255) / 256, 256, 0, stream>>>(x, xb);
  prep_w1<<<(32 * 192 + 255) / 256, 256, 0, stream>>>(W1, w1b);
  prep_w2<<<(192 * 48 + 255) / 256, 256, 0, stream>>>(W2, w2b);
  prep_wT<<<(96 * 192 + 255) / 256, 256, 0, stream>>>(Wih0, wih0b, 96);
  prep_wT<<<(64 * 192 + 255) / 256, 256, 0, stream>>>(Whh0, whh0b, 64);
  prep_wT<<<(64 * 192 + 255) / 256, 256, 0, stream>>>(Wih1, wih1b, 64);
  prep_wT<<<(64 * 192 + 255) / 256, 256, 0, stream>>>(Whh1, whh1b, 64);

  // GAT layer 1
  gemm1<<<(1472 * 12) / 8, 256, 0, stream>>>(xb, w1b, h1b);
  al1_kernel<<<(NTOT * 4 + 255) / 256, 256, 0, stream>>>(h1b, as1, ad1, al1s, al1d);
  agg1<<<NG, 256, 0, stream>>>(h1b, al1s, al1d, b1, act1b);

  // GAT layer 2 + pooling
  gemm2<<<(1472 * 3) / 8, 256, 0, stream>>>(act1b, w2b, h2b);
  al2_kernel<<<(NTOT + 255) / 256, 256, 0, stream>>>(h2b, as2, ad2, al2s, al2d);
  agg2_pool<<<NG, 256, 0, stream>>>(h2b, al2s, al2d, b2, framesb);

  // GRU stack (persistent single-WGP kernels; sequential over T inside)
  gru_layer<<<1, 256, 0, stream>>>(framesb, 3, wih0b, whh0b, bih0, bhh0, l0seq, nullptr);
  gru_layer<<<1, 256, 0, stream>>>(l0seq, 2, wih1b, whh1b, bih1, bhh1, nullptr, last);

  // MLP head
  head_kernel<<<1, 64, 0, stream>>>(last, Wh1, bh1, Wh2, bh2, out);
}
